// GPTBlock_56650618634901
// MI455X (gfx1250) — matrix operand order repaired and checked by reading, not compiled, not run on hardware
//
#include <hip/hip_runtime.h>

// ---------------------------------------------------------------------------
// Types / helpers for CDNA5 WMMA (gfx1250, wave32)
// ---------------------------------------------------------------------------
typedef __attribute__((ext_vector_type(16))) __bf16 v16bf;
typedef __attribute__((ext_vector_type(8)))  float  v8f;
typedef __attribute__((ext_vector_type(4)))  unsigned int u32x4;
typedef __attribute__((ext_vector_type(4)))  int i32x4;
typedef __attribute__((ext_vector_type(8)))  int i32x8;

union Frag16 {
  v16bf v;
  unsigned short u[16];
  unsigned int   d[8];
};

union U4 {
  uint4 q;
  unsigned short u[8];
};

// Native bf16 conversion (hardware cvt on gfx1250)
__device__ __forceinline__ unsigned short f2bf(float f) {
  union { __bf16 h; unsigned short u; } cv;
  cv.h = (__bf16)f;
  return cv.u;
}

__device__ __forceinline__ unsigned pk2bf(float lo, float hi) {
  return (unsigned)f2bf(lo) | ((unsigned)f2bf(hi) << 16);
}

__device__ __forceinline__ v8f wmma_bf16(const Frag16& a, const Frag16& b, v8f c) {
  // D = A(16x32 bf16) * B(32x16 bf16) + C(16x16 f32)
  return __builtin_amdgcn_wmma_f32_16x16x32_bf16(
      /*neg_a=*/false, a.v, /*neg_b=*/false, b.v,
      /*c_mod=*/(short)0, c, /*reuse_a=*/false, /*reuse_b=*/false);
}

// ---------------------------------------------------------------------------
// Tensor Data Mover: 2D tile (bf16, data_size=2B) global -> LDS.
// Descriptor bit layout per CDNA5 ISA ch.8 (D# group0/group1).
// 6-arg builtin (clang-23 / therock-10.0): (g0, g1, g2, g3, g4, cpol).
// ---------------------------------------------------------------------------
__device__ __forceinline__ void tdm_load_2d_bf16(
    unsigned lds_off, const void* gaddr, unsigned tile_d0, unsigned tile_d1,
    unsigned tensor_d0, unsigned tensor_d1, unsigned stride_d0) {
  unsigned long long ga = (unsigned long long)(size_t)gaddr;
  u32x4 g0;
  g0[0] = 1u;                                  // count=1 (valid), user mode
  g0[1] = lds_off;                             // LDS byte address
  g0[2] = (unsigned)(ga & 0xFFFFFFFFu);        // global_addr[31:0]
  g0[3] = (unsigned)((ga >> 32) & 0x01FFFFFFu) | (2u << 30);  // addr hi | type=2
  i32x8 g1;
  g1[0] = (int)(1u << 16);                     // wg_mask=0, data_size=1 (2B)
  g1[1] = (int)((tensor_d0 & 0xFFFFu) << 16);  // tensor_dim0[15:0]
  g1[2] = (int)(((tensor_d0 >> 16) & 0xFFFFu) | ((tensor_d1 & 0xFFFFu) << 16));
  g1[3] = (int)(((tensor_d1 >> 16) & 0xFFFFu) | ((tile_d0 & 0xFFFFu) << 16));
  g1[4] = (int)(tile_d1 & 0xFFFFu);            // tile_dim1 (tile_dim2 = 0)
  g1[5] = (int)stride_d0;                      // tensor_dim0_stride[31:0]
  g1[6] = 0;
  g1[7] = 0;
  i32x4 gz4 = {0, 0, 0, 0};
  i32x8 gz8 = {0, 0, 0, 0, 0, 0, 0, 0};
  __builtin_amdgcn_tensor_load_to_lds(g0, g1, gz4, gz4, gz8, 0);
}

// ---------------------------------------------------------------------------
// fp32 -> bf16 conversion, 8 elements per thread (all sizes are mult. of 8)
// ---------------------------------------------------------------------------
__global__ void cvt_f32_to_bf16_v8(const float* __restrict__ in,
                                   unsigned short* __restrict__ out, int n8) {
  int i = blockIdx.x * blockDim.x + threadIdx.x;
  if (i < n8) {
    const float4* p = reinterpret_cast<const float4*>(in + (long)i * 8);
    float4 a = p[0];
    float4 b = p[1];
    u32x4 o;
    o[0] = pk2bf(a.x, a.y);
    o[1] = pk2bf(a.z, a.w);
    o[2] = pk2bf(b.x, b.y);
    o[3] = pk2bf(b.z, b.w);
    *reinterpret_cast<u32x4*>(out + (long)i * 8) = o;
  }
}

// ---------------------------------------------------------------------------
// Tiled WMMA GEMM: C[M,N] = epi(A[M,K] * B[K,N] + bias)
// A,B bf16 row-major.  256 threads = 8 waves; tile 128x128; each wave owns a
// 64x32 sub-tile (4x2 WMMA tiles).  BK = 64.
//   - A tile staged row-major in LDS via the Tensor Data Mover (wave 0 issues
//     tensor_load_to_lds, waits TENSORcnt, barrier releases all waves).
//   - B tile staged pair-interleaved: (k,col) -> Bs[(k>>1)*256 + col*2 + (k&1)]
//     so each B-fragment dword is a single aligned b32 ds load.
// EPI: 0 = store bf16; 1 = bias+relu bf16; 2 = bias f32.
// ---------------------------------------------------------------------------
template <int EPI>
__global__ __launch_bounds__(256) void gemm_bf16_wmma(
    const unsigned short* __restrict__ A, const unsigned short* __restrict__ Bm,
    const float* __restrict__ bias, void* __restrict__ Cout,
    int M, int N, int K) {
  __shared__ unsigned short As[128 * 64];  // 16 KB, row-major (TDM dest)
  __shared__ unsigned short Bs[32 * 256];  // 16 KB, pair-interleaved

  const int tid  = threadIdx.x;
  const int wave = tid >> 5;
  const int lane = tid & 31;
  const int wm   = wave >> 2;       // 0..1  -> row offset 64*wm
  const int wn   = wave & 3;        // 0..3  -> col offset 32*wn
  const int half = lane >> 4;
  const int l15  = lane & 15;
  const long rowBase = (long)blockIdx.y * 128;
  const int  colBase = blockIdx.x * 128;

  const v8f zero8 = {0.f, 0.f, 0.f, 0.f, 0.f, 0.f, 0.f, 0.f};
  v8f acc[4][2];
#pragma unroll
  for (int mi = 0; mi < 4; ++mi)
#pragma unroll
    for (int ni = 0; ni < 2; ++ni) acc[mi][ni] = zero8;

  for (int k0 = 0; k0 < K; k0 += 64) {
    // ---- A tile via TDM (wave 0 only; TENSORcnt wait, then barrier) ----
    if (wave == 0) {
      tdm_load_2d_bf16((unsigned)(size_t)(void*)As,
                       (const void*)(A + rowBase * K + k0),
                       /*tile_d0=*/64, /*tile_d1=*/128,
                       /*tensor_d0=*/(unsigned)K, /*tensor_d1=*/(unsigned)M,
                       /*stride_d0=*/(unsigned)K);
      __builtin_amdgcn_s_wait_tensorcnt(0);
    }
    // ---- B tile, pair-interleaved (512 pair-chunks, 2/thread) ----
#pragma unroll
    for (int c = 0; c < 2; ++c) {
      int pc   = tid + c * 256;        // 0..511
      int k2   = pc >> 4;              // 0..31 row pair
      int col8 = (pc & 15) * 8;        // 0..120
      U4 lo, hi, o0, o1;
      lo.q = *reinterpret_cast<const uint4*>(Bm + (long)(k0 + 2 * k2) * N + colBase + col8);
      hi.q = *reinterpret_cast<const uint4*>(Bm + (long)(k0 + 2 * k2 + 1) * N + colBase + col8);
#pragma unroll
      for (int e = 0; e < 4; ++e) { o0.u[2 * e] = lo.u[e];     o0.u[2 * e + 1] = hi.u[e]; }
#pragma unroll
      for (int e = 0; e < 4; ++e) { o1.u[2 * e] = lo.u[e + 4]; o1.u[2 * e + 1] = hi.u[e + 4]; }
      *reinterpret_cast<uint4*>(&Bs[k2 * 256 + col8 * 2])     = o0.q;
      *reinterpret_cast<uint4*>(&Bs[k2 * 256 + col8 * 2 + 8]) = o1.q;
    }
    __syncthreads();

#pragma unroll
    for (int ks = 0; ks < 64; ks += 32) {
      Frag16 af[4];
#pragma unroll
      for (int mi = 0; mi < 4; ++mi) {
        int row = wm * 64 + mi * 16 + l15;
#pragma unroll
        for (int i = 0; i < 8; ++i) {
          int kk = ks + half * 8 + ((i < 4) ? 2 * i : 2 * i + 8);
          af[mi].d[i] = *reinterpret_cast<const unsigned int*>(&As[row * 64 + kk]);
        }
      }
      Frag16 bf[2];
#pragma unroll
      for (int ni = 0; ni < 2; ++ni) {
        int col = wn * 32 + ni * 16 + l15;
#pragma unroll
        for (int j = 0; j < 8; ++j) {
          int kp = (ks >> 1) + half * 4 + ((j < 4) ? j : j + 4);   // pair index (k>>1)
          bf[ni].d[j] = *reinterpret_cast<const unsigned int*>(&Bs[kp * 256 + col * 2]);
        }
      }
#pragma unroll
      for (int mi = 0; mi < 4; ++mi)
#pragma unroll
        for (int ni = 0; ni < 2; ++ni)
          acc[mi][ni] = wmma_bf16(af[mi], bf[ni], acc[mi][ni]);
    }
    __syncthreads();
  }

  // Epilogue.  C layout: vgpr r -> M = r + 8*half; N = lane&15.
#pragma unroll
  for (int mi = 0; mi < 4; ++mi) {
#pragma unroll
    for (int ni = 0; ni < 2; ++ni) {
      int gcol = colBase + wn * 32 + ni * 16 + l15;
      float bv = (EPI >= 1) ? bias[gcol] : 0.f;
#pragma unroll
      for (int r = 0; r < 8; ++r) {
        long grow = rowBase + wm * 64 + mi * 16 + r + 8 * half;
        float val = acc[mi][ni][r] + bv;
        if (EPI == 1) val = fmaxf(val, 0.f);
        if (EPI == 2)
          reinterpret_cast<float*>(Cout)[grow * N + gcol] = val;
        else
          reinterpret_cast<unsigned short*>(Cout)[grow * N + gcol] = f2bf(val);
      }
    }
  }
}

// ---------------------------------------------------------------------------
// Causal flash attention (single head, d = 512, no 1/sqrt(d) scaling).
// One workgroup per (batch, 32-row Q tile).  8 waves split d into 64-wide
// slices.  Partial 32x32 scores reduce via ds_add_f32 into one LDS buffer;
// online softmax on one wave; V tile staged pair-interleaved in LDS so PV
// B-fragments are single b32 ds loads.
// ---------------------------------------------------------------------------
__global__ __launch_bounds__(256) void flash_attn_causal(
    const unsigned short* __restrict__ Q, const unsigned short* __restrict__ Kt,
    const unsigned short* __restrict__ Vt, unsigned short* __restrict__ Onew) {
  const int T = 4096, C = 512;
  const int qt    = blockIdx.x;       // 0..127
  const int bidx  = blockIdx.y;       // 0..3
  const int qbase = qt * 32;
  const long base = (long)bidx * T * C;
  const int tid  = threadIdx.x;
  const int wave = tid >> 5;
  const int lane = tid & 31;
  const int half = lane >> 4;
  const int l15  = lane & 15;
  const int cslice = wave * 64;

  __shared__ float          S1[32][32];       // reduced scores (4 KB)
  __shared__ unsigned short Vs[16 * 1024];    // V tile, pair-interleaved (32 KB)
  __shared__ unsigned short Pt[32][32];       // probs bf16 (2 KB)
  __shared__ float          rmax[32], rsum[32], rscale[32];

  const v8f zero8 = {0.f, 0.f, 0.f, 0.f, 0.f, 0.f, 0.f, 0.f};
  v8f acc[2][4];
#pragma unroll
  for (int mi = 0; mi < 2; ++mi)
#pragma unroll
    for (int ni = 0; ni < 4; ++ni) acc[mi][ni] = zero8;

  if (tid < 32) { rmax[tid] = -1e30f; rsum[tid] = 0.f; }
  __syncthreads();

  for (int kt = 0; kt <= qt; ++kt) {
    const int kbase = kt * 32;

    // ---- zero S + stage V tile pair-interleaved (1024 pair-chunks) ----
    {
      float* Sf = &S1[0][0];
#pragma unroll
      for (int e = 0; e < 4; ++e) Sf[tid * 4 + e] = 0.f;
    }
#pragma unroll
    for (int c = 0; c < 4; ++c) {
      int pc = tid + c * 256;          // 0..1023
      int k2 = pc >> 6;                // 0..15 row pair
      int c8 = (pc & 63) * 8;          // 0..504
      U4 lo, hi, o0, o1;
      lo.q = *reinterpret_cast<const uint4*>(Vt + base + (long)(kbase + 2 * k2) * C + c8);
      hi.q = *reinterpret_cast<const uint4*>(Vt + base + (long)(kbase + 2 * k2 + 1) * C + c8);
#pragma unroll
      for (int e = 0; e < 4; ++e) { o0.u[2 * e] = lo.u[e];     o0.u[2 * e + 1] = hi.u[e]; }
#pragma unroll
      for (int e = 0; e < 4; ++e) { o1.u[2 * e] = lo.u[e + 4]; o1.u[2 * e + 1] = hi.u[e + 4]; }
      *reinterpret_cast<uint4*>(&Vs[k2 * 1024 + c8 * 2])     = o0.q;
      *reinterpret_cast<uint4*>(&Vs[k2 * 1024 + c8 * 2 + 8]) = o1.q;
    }
    // prefetch next K tile while we compute on this one
    if (kt < qt) {
      const unsigned short* pf =
          Kt + base + (long)(kbase + 32 + (tid & 31)) * C + (tid >> 5) * 64;
      __builtin_prefetch(pf, 0, 1);
    }
    __syncthreads();

    // ---- partial S = Q[32, cslice] * K[32, cslice]^T over this c-slice ----
    v8f s[2][2];
#pragma unroll
    for (int mi = 0; mi < 2; ++mi)
#pragma unroll
      for (int ni = 0; ni < 2; ++ni) s[mi][ni] = zero8;

#pragma unroll
    for (int ks = 0; ks < 64; ks += 32) {
      Frag16 a[2];
#pragma unroll
      for (int mi = 0; mi < 2; ++mi) {
        int row = qbase + mi * 16 + l15;
        const unsigned short* qp = Q + base + (long)row * C + cslice + ks;
#pragma unroll
        for (int i = 0; i < 8; ++i) {
          int kk = half * 8 + ((i < 4) ? 2 * i : 2 * i + 8);
          a[mi].d[i] = *reinterpret_cast<const unsigned int*>(qp + kk);
        }
      }
      Frag16 bb[2];
#pragma unroll
      for (int ni = 0; ni < 2; ++ni) {
        int key = kbase + ni * 16 + l15;
        const unsigned short* kp = Kt + base + (long)key * C + cslice + ks;
#pragma unroll
        for (int j = 0; j < 8; ++j) {
          int kk = half * 8 + ((j < 4) ? 2 * j : 2 * j + 8);  // table-order pair along c
          bb[ni].d[j] = *reinterpret_cast<const unsigned int*>(kp + kk);
        }
      }
#pragma unroll
      for (int mi = 0; mi < 2; ++mi)
#pragma unroll
        for (int ni = 0; ni < 2; ++ni)
          s[mi][ni] = wmma_bf16(a[mi], bb[ni], s[mi][ni]);
    }

    // ---- reduce partials across waves via LDS float atomics ----
#pragma unroll
    for (int mi = 0; mi < 2; ++mi)
#pragma unroll
      for (int ni = 0; ni < 2; ++ni)
#pragma unroll
        for (int r = 0; r < 8; ++r)
          atomicAdd(&S1[mi * 16 + r + 8 * half][ni * 16 + l15], s[mi][ni][r]);
    __syncthreads();

    // ---- online softmax (one lane per query row), causal mask in-read ----
    if (tid < 32) {
      int m = tid;
      float sc[32];
#pragma unroll
      for (int n = 0; n < 32; ++n)
        sc[n] = (kbase + n > qbase + m) ? -1e30f : S1[m][n];
      float mx = rmax[m];
      float tm = -1e30f;
#pragma unroll
      for (int n = 0; n < 32; ++n) tm = fmaxf(tm, sc[n]);
      float mn    = fmaxf(mx, tm);
      float alpha = __expf(mx - mn);
      float sum   = 0.f;
#pragma unroll
      for (int n = 0; n < 32; n += 2) {
        float p0 = __expf(sc[n] - mn);
        float p1 = __expf(sc[n + 1] - mn);
        sum += p0 + p1;
        *reinterpret_cast<unsigned*>(&Pt[m][n]) = pk2bf(p0, p1);
      }
      rsum[m]   = rsum[m] * alpha + sum;
      rmax[m]   = mn;
      rscale[m] = alpha;
    }
    __syncthreads();

    // ---- rescale accumulators by alpha[row] ----
#pragma unroll
    for (int mi = 0; mi < 2; ++mi)
#pragma unroll
      for (int r = 0; r < 8; ++r) {
        float al = rscale[mi * 16 + r + 8 * half];
#pragma unroll
        for (int ni = 0; ni < 4; ++ni) acc[mi][ni][r] *= al;
      }

    // ---- O += P(32x32) * V(32 x 64-slice), V from pair-interleaved LDS ----
    Frag16 pa[2];
#pragma unroll
    for (int mi = 0; mi < 2; ++mi) {
      int m = mi * 16 + l15;
#pragma unroll
      for (int i = 0; i < 8; ++i) {
        int kk = half * 8 + ((i < 4) ? 2 * i : 2 * i + 8);
        pa[mi].d[i] = *reinterpret_cast<const unsigned int*>(&Pt[m][kk]);
      }
    }
#pragma unroll
    for (int ni = 0; ni < 4; ++ni) {
      Frag16 vb;
      int c = cslice + ni * 16 + l15;
#pragma unroll
      for (int j = 0; j < 8; ++j) {
        int kp = half * 4 + ((j < 4) ? j : j + 4);  // pair index within 32-key tile
        vb.d[j] = *reinterpret_cast<const unsigned int*>(&Vs[kp * 1024 + c * 2]);
      }
#pragma unroll
      for (int mi = 0; mi < 2; ++mi)
        acc[mi][ni] = wmma_bf16(pa[mi], vb, acc[mi][ni]);
    }
    __syncthreads();
  }

  // ---- finalize: O /= rowsum, store bf16 ----
#pragma unroll
  for (int mi = 0; mi < 2; ++mi)
#pragma unroll
    for (int r = 0; r < 8; ++r) {
      int m = mi * 16 + r + 8 * half;
      float inv = 1.f / rsum[m];
#pragma unroll
      for (int ni = 0; ni < 4; ++ni) {
        int gc = cslice + ni * 16 + l15;
        Onew[base + (long)(qbase + m) * C + gc] = f2bf(acc[mi][ni][r] * inv);
      }
    }
}

// ---------------------------------------------------------------------------
// Host orchestration
// ---------------------------------------------------------------------------
extern "C" void kernel_launch(void* const* d_in, const int* in_sizes, int n_in,
                              void* d_out, int out_size, void* d_ws, size_t ws_size,
                              hipStream_t stream) {
  const int Bc = 4, T = 4096, C = 512, DFF = 2048;
  const long M = (long)Bc * T;  // 16384

  const float* X  = (const float*)d_in[0];
  const float* Wq = (const float*)d_in[1];
  const float* Wk = (const float*)d_in[2];
  const float* Wv = (const float*)d_in[3];
  const float* W1 = (const float*)d_in[4];
  const float* b1 = (const float*)d_in[5];
  const float* W2 = (const float*)d_in[6];
  const float* b2 = (const float*)d_in[7];

  char* ws = (char*)d_ws;
  size_t off = 0;
  auto take = [&](size_t bytes) -> char* {
    char* p = ws + off;
    off = (off + bytes + 255) & ~(size_t)255;
    return p;
  };

  unsigned short* Xb  = (unsigned short*)take((size_t)M * C * 2);
  unsigned short* Wqb = (unsigned short*)take((size_t)C * C * 2);
  unsigned short* Wkb = (unsigned short*)take((size_t)C * C * 2);
  unsigned short* Wvb = (unsigned short*)take((size_t)C * C * 2);
  unsigned short* W1b = (unsigned short*)take((size_t)C * DFF * 2);
  unsigned short* W2b = (unsigned short*)take((size_t)DFF * C * 2);
  unsigned short* Qb  = (unsigned short*)take((size_t)M * C * 2);
  unsigned short* Kb  = (unsigned short*)take((size_t)M * C * 2);
  unsigned short* Vb  = (unsigned short*)take((size_t)M * C * 2);
  unsigned short* NV  = (unsigned short*)take((size_t)M * C * 2);
  unsigned short* Hb  = (unsigned short*)take((size_t)M * DFF * 2);

  auto cvt = [&](const float* src, unsigned short* dst, long n) {
    long n8 = n / 8;
    cvt_f32_to_bf16_v8<<<dim3((unsigned)((n8 + 255) / 256)), dim3(256), 0, stream>>>(
        src, dst, (int)n8);
  };
  cvt(X, Xb, M * C);
  cvt(Wq, Wqb, (long)C * C);
  cvt(Wk, Wkb, (long)C * C);
  cvt(Wv, Wvb, (long)C * C);
  cvt(W1, W1b, (long)C * DFF);
  cvt(W2, W2b, (long)DFF * C);

  dim3 blk(256);
  // Q/K/V projections: [16384,512] x [512,512] -> bf16
  gemm_bf16_wmma<0><<<dim3(C / 128, (unsigned)(M / 128)), blk, 0, stream>>>(
      Xb, Wqb, nullptr, Qb, (int)M, C, C);
  gemm_bf16_wmma<0><<<dim3(C / 128, (unsigned)(M / 128)), blk, 0, stream>>>(
      Xb, Wkb, nullptr, Kb, (int)M, C, C);
  gemm_bf16_wmma<0><<<dim3(C / 128, (unsigned)(M / 128)), blk, 0, stream>>>(
      Xb, Wvb, nullptr, Vb, (int)M, C, C);

  // Causal attention -> newV (bf16)
  flash_attn_causal<<<dim3(T / 32, Bc), blk, 0, stream>>>(Qb, Kb, Vb, NV);

  // MLP: H = relu(newV @ W1 + b1)  (bf16 out)
  gemm_bf16_wmma<1><<<dim3(DFF / 128, (unsigned)(M / 128)), blk, 0, stream>>>(
      NV, W1b, b1, Hb, (int)M, DFF, C);
  // out = H @ W2 + b2  (fp32 out)
  gemm_bf16_wmma<2><<<dim3(C / 128, (unsigned)(M / 128)), blk, 0, stream>>>(
      Hb, W2b, b2, d_out, (int)M, C, DFF);
}
